// TabM_40321152975418
// MI455X (gfx1250) — matrix operand order repaired and checked by reading, not compiled, not run on hardware
//
#include <hip/hip_runtime.h>
#include <cstdint>
#include <cstddef>

// ---------------- CDNA5 WMMA types ----------------
typedef __attribute__((ext_vector_type(16))) __bf16 v16bf;
typedef __attribute__((ext_vector_type(8)))  float  v8f;

union FragBF {
    v16bf  v;
    __bf16 h[16];
    uint4  q[2];
};

// ---------------- problem constants ----------------
#define N_MEMBERS 32
#define IN_DIM    512
#define HID       512
#define HID2      256
#define BATCH_N   8192
#define MTILE     64            // batch rows per workgroup
#define XSTR      520           // __bf16 stride of x/h1 rows (260-dword stride -> conflict-free b128 A loads)
#define H2STR     260           // float stride of h2 rows (aliases XSTR bytes exactly)

// B fragment (32x16 bf16) gathered directly from global f32 weights.
// ISA B layout: lanes 0-15 hold K=0..7,16..23 (dword i = K pair), lanes 16-31 hold K=8..15,24..31.
template <int N>
__device__ __forceinline__ void load_bfrag(FragBF& f, const float* __restrict__ Wg,
                                           int kc, int col, int k16)
{
    #pragma unroll
    for (int i = 0; i < 8; ++i) {
        const float* p = Wg + (size_t)(kc + 2 * i + ((i & 4) << 1) + (k16 >> 1)) * N + col;
        f.h[2 * i]     = (__bf16)p[0];
        f.h[2 * i + 1] = (__bf16)p[N];
    }
}

// A fragment (16x32 bf16) from the LDS activation tile.
// ISA A layout: dwords 0-3 = K kc+{0,2,4,6}+lhalf8 (contiguous 16B), dwords 4-7 = +16.
__device__ __forceinline__ void load_afrag(FragBF& f, const __bf16* __restrict__ rowp,
                                           int kc, int lhalf8)
{
    f.q[0] = *(const uint4*)&rowp[kc + lhalf8];
    f.q[1] = *(const uint4*)&rowp[kc + 16 + lhalf8];
}

union Bf4 { __bf16 h[4]; unsigned long long ll; };

// Fused 3-layer ensemble member MLP. One workgroup = one (member, 64-row batch tile).
__global__ __launch_bounds__(256) void tabm_fused(
    const float* __restrict__ x,  const float* __restrict__ W1, const float* __restrict__ b1,
    const float* __restrict__ W2, const float* __restrict__ b2,
    const float* __restrict__ W3, const float* __restrict__ b3,
    float* __restrict__ preds /* [N_MEMBERS][BATCH] */)
{
    extern __shared__ unsigned char smem[];
    __bf16* xh  = (__bf16*)smem;     // [MTILE][XSTR]  x-tile, later h1
    float*  h2f = (float*)smem;      // alias: [MTILE][H2STR] h2 (f32)

    const int tid  = threadIdx.x;
    const int lane = tid & 31;
    const int wave = tid >> 5;
    const int m    = blockIdx.y;
    const int row0 = blockIdx.x * MTILE;

    const int lm     = lane & 15;         // M (A/C) or N (B/C) index within a 16-lane half
    const int lhalf8 = (lane & 16) >> 1;  // +8 (A-K / C-M) for upper lane half
    const int k16    = lane & 16;         // +16 (B-K) for upper lane half

    // ---- stage x tile: fp32 global -> bf16 LDS (float4 load, 8B packed store) ----
    for (int u4 = tid; u4 < MTILE * IN_DIM / 4; u4 += 256) {
        int r = u4 >> 7;                  // 128 quads per 512-wide row
        int c = (u4 & 127) * 4;
        const float4 f = *(const float4*)&x[(size_t)(row0 + r) * IN_DIM + c];
        Bf4 b;
        b.h[0] = (__bf16)f.x; b.h[1] = (__bf16)f.y;
        b.h[2] = (__bf16)f.z; b.h[3] = (__bf16)f.w;
        *(unsigned long long*)&xh[r * XSTR + c] = b.ll;
    }
    __syncthreads();

    // ================= layer 1: h1 = relu(x @ W1[m] + b1[m]) =================
    const float* W1g = W1 + (size_t)m * IN_DIM * HID;
    v8f acc[4][4];
    {
        v8f z = {0.f, 0.f, 0.f, 0.f, 0.f, 0.f, 0.f, 0.f};
        for (int i = 0; i < 4; ++i)
            for (int j = 0; j < 4; ++j) acc[i][j] = z;
    }
    const int wn0 = wave * 64;            // this wave's 64 output columns (disjoint per wave)

    for (int kc = 0; kc < IN_DIM; kc += 32) {
        FragBF a[4];
        #pragma unroll
        for (int mt = 0; mt < 4; ++mt)
            load_afrag(a[mt], &xh[(mt * 16 + lm) * XSTR], kc, lhalf8);
        #pragma unroll
        for (int nt = 0; nt < 4; ++nt) {
            const int col = wn0 + nt * 16 + lm;
            if (kc + 32 < IN_DIM)
                __builtin_prefetch(&W1g[(size_t)(kc + 32 + k16) * HID + col], 0, 1);
            FragBF bfr;
            load_bfrag<HID>(bfr, W1g, kc, col, k16);
            #pragma unroll
            for (int mt = 0; mt < 4; ++mt)
                acc[mt][nt] = __builtin_amdgcn_wmma_f32_16x16x32_bf16(
                    false, a[mt].v, false, bfr.v, (short)0, acc[mt][nt], false, false);
        }
    }

    __syncthreads();   // everyone done reading x-tile
    // epilogue: bias + relu, write h1 (bf16) over the x-tile region (each wave owns its cols)
    #pragma unroll
    for (int nt = 0; nt < 4; ++nt) {
        int col = wn0 + nt * 16 + lm;
        float bias = b1[(size_t)m * HID + col];
        #pragma unroll
        for (int mt = 0; mt < 4; ++mt) {
            #pragma unroll
            for (int v = 0; v < 8; ++v) {
                int r = mt * 16 + v + lhalf8;
                xh[r * XSTR + col] = (__bf16)fmaxf(acc[mt][nt][v] + bias, 0.f);
            }
        }
    }
    __syncthreads();

    // ================= layer 2: h2 = relu(h1 @ W2[m] + b2[m]) =================
    const float* W2g = W2 + (size_t)m * HID * HID2;
    v8f acc2[4][2];
    {
        v8f z = {0.f, 0.f, 0.f, 0.f, 0.f, 0.f, 0.f, 0.f};
        for (int i = 0; i < 4; ++i)
            for (int j = 0; j < 2; ++j) acc2[i][j] = z;
    }
    const int wn2 = wave * 32;

    for (int kc = 0; kc < HID; kc += 32) {
        FragBF a[4];
        #pragma unroll
        for (int mt = 0; mt < 4; ++mt)
            load_afrag(a[mt], &xh[(mt * 16 + lm) * XSTR], kc, lhalf8);
        #pragma unroll
        for (int nt = 0; nt < 2; ++nt) {
            const int col = wn2 + nt * 16 + lm;
            if (kc + 32 < HID)
                __builtin_prefetch(&W2g[(size_t)(kc + 32 + k16) * HID2 + col], 0, 1);
            FragBF bfr;
            load_bfrag<HID2>(bfr, W2g, kc, col, k16);
            #pragma unroll
            for (int mt = 0; mt < 4; ++mt)
                acc2[mt][nt] = __builtin_amdgcn_wmma_f32_16x16x32_bf16(
                    false, a[mt].v, false, bfr.v, (short)0, acc2[mt][nt], false, false);
        }
    }

    __syncthreads();   // everyone done reading h1
    // epilogue: bias + relu, write h2 (f32) over the tile region
    #pragma unroll
    for (int nt = 0; nt < 2; ++nt) {
        int col = wn2 + nt * 16 + lm;
        float bias = b2[(size_t)m * HID2 + col];
        #pragma unroll
        for (int mt = 0; mt < 4; ++mt) {
            #pragma unroll
            for (int v = 0; v < 8; ++v) {
                int r = mt * 16 + v + lhalf8;
                h2f[r * H2STR + col] = fmaxf(acc2[mt][nt][v] + bias, 0.f);
            }
        }
    }
    __syncthreads();

    // ================= layer 3: preds = h2 @ W3[m] + b3[m] =================
    if (tid < MTILE) {
        const float* W3g = W3 + (size_t)m * HID2;
        float s = 0.f;
        for (int c = 0; c < HID2; ++c) s += h2f[tid * H2STR + c] * W3g[c];
        preds[(size_t)m * BATCH_N + row0 + tid] = s + b3[m];
    }
}

// Deterministic member combine: out[b] = bc + sum_m preds[m][b] * Wc[m]
__global__ __launch_bounds__(256) void tabm_combine(
    const float* __restrict__ preds, const float* __restrict__ Wc,
    const float* __restrict__ bc, float* __restrict__ out)
{
    int b = blockIdx.x * 256 + threadIdx.x;
    if (b >= BATCH_N) return;
    float s = bc[0];
    #pragma unroll
    for (int mm = 0; mm < N_MEMBERS; ++mm)
        s += preds[(size_t)mm * BATCH_N + b] * Wc[mm];
    out[b] = s;
}

extern "C" void kernel_launch(void* const* d_in, const int* in_sizes, int n_in,
                              void* d_out, int out_size, void* d_ws, size_t ws_size,
                              hipStream_t stream)
{
    (void)in_sizes; (void)n_in; (void)out_size; (void)ws_size;
    const float* x  = (const float*)d_in[0];
    const float* W1 = (const float*)d_in[1];
    const float* b1 = (const float*)d_in[2];
    const float* W2 = (const float*)d_in[3];
    const float* b2 = (const float*)d_in[4];
    const float* W3 = (const float*)d_in[5];
    const float* b3 = (const float*)d_in[6];
    const float* Wc = (const float*)d_in[7];
    const float* bc = (const float*)d_in[8];

    float* preds = (float*)d_ws;    // 32 * 8192 floats = 1 MB scratch

    dim3 grid(BATCH_N / MTILE, N_MEMBERS);
    size_t smem = (size_t)MTILE * XSTR * sizeof(__bf16);   // 66560 B: activation tile only

    tabm_fused<<<grid, 256, smem, stream>>>(x, W1, b1, W2, b2, W3, b3, preds);
    tabm_combine<<<BATCH_N / 256, 256, 0, stream>>>(preds, Wc, bc, (float*)d_out);
}
